// EncoderLayer_48515950576095
// MI455X (gfx1250) — matrix operand order repaired and checked by reading, not compiled, not run on hardware
//
#include <hip/hip_runtime.h>
#include <hip/hip_bf16.h>

// ---------------------------------------------------------------------------
// Types
// ---------------------------------------------------------------------------
typedef _Float16 half_t;
typedef _Float16 v16h __attribute__((ext_vector_type(16)));
typedef _Float16 v8h  __attribute__((ext_vector_type(8)));
typedef float    v8f  __attribute__((ext_vector_type(8)));

#define DEV __device__ __attribute__((always_inline)) inline

DEV v8f wmma_f16(v16h a, v16h b, v8f c) {
  // (neg_a, A, neg_b, B, c_mod, C, reuse_a, reuse_b)
  return __builtin_amdgcn_wmma_f32_16x16x32_f16(false, a, false, b, (short)0, c,
                                                false, false);
}

DEV v16h ld_frag(const half_t* p0, const half_t* p1) {
  union { v16h v; v8h h[2]; } u;
  u.h[0] = *(const v8h*)p0;
  u.h[1] = *(const v8h*)p1;
  return u.v;
}

// ---------------------------------------------------------------------------
// Problem constants
// ---------------------------------------------------------------------------
#define CB 4
#define CS 2048
#define CD 512
#define CH 8
#define CHD 64
#define CF 2048
#define CM (CB * CS)   // 8192 rows

// ---------------------------------------------------------------------------
// Optional CDNA5 Tensor Data Mover staging path (guarded, with fallback)
// ---------------------------------------------------------------------------
#if defined(__gfx1250__) && __has_builtin(__builtin_amdgcn_tensor_load_to_lds)
#define USE_TDM 1
#endif

#ifdef USE_TDM
typedef unsigned int u32x4 __attribute__((ext_vector_type(4)));
typedef int          i32x4 __attribute__((ext_vector_type(4)));
typedef int          i32x8 __attribute__((ext_vector_type(8)));

#if __has_builtin(__builtin_amdgcn_s_wait_tensorcnt)
#define WAIT_TENSOR() __builtin_amdgcn_s_wait_tensorcnt(0)
#else
#define WAIT_TENSOR() asm volatile("s_wait_tensorcnt 0x0" ::: "memory")
#endif

// 2D tile load: tile_d0 x tile_d1 elements of 2 bytes, row stride = stride0
// elements, LDS destination padded by 4 DWORDs every 16 DWORDs (i.e. row pitch
// 32 halves -> 40 halves, matching LDA/LDB below).
DEV void tdm_load_2d(unsigned lds_addr, unsigned long long gaddr,
                     unsigned tensor_d0, unsigned tensor_d1,
                     unsigned long long stride0, unsigned tile_d0,
                     unsigned tile_d1) {
  union { unsigned long long q[2]; u32x4 v; } g0;
  // count=1 (valid), gather off; lds_addr [63:32]; global_addr [120:64];
  // type=2 ("image") in [127:126]
  g0.q[0] = 1ull | ((unsigned long long)lds_addr << 32);
  g0.q[1] = (gaddr & ((1ull << 57) - 1ull)) | (2ull << 62);

  union { unsigned long long q[4]; i32x8 v; } g1;
  // workgroup_mask=0; data_size=1 (2B) at [17:16]; pad_enable at [20];
  // pad_interval=3 (16 DWORDs) at [24:22]; pad_amount=3 (4 DWORDs) at [31:25]
  unsigned long long flags =
      (1ull << 16) | (1ull << 20) | (3ull << 22) | (3ull << 25);
  g1.q[0] = flags | ((unsigned long long)(tensor_d0 & 0xFFFFu) << 48);
  g1.q[1] = (unsigned long long)(tensor_d0 >> 16) |
            ((unsigned long long)tensor_d1 << 16) |
            ((unsigned long long)(tile_d0 & 0xFFFFu) << 48);
  g1.q[2] = (unsigned long long)(tile_d1 & 0xFFFFu) |
            ((stride0 & 0xFFFFFFFFull) << 32);
  g1.q[3] = (stride0 >> 32) & 0xFFFFull;  // dim1_stride unused for 2D tiles

  i32x4 z4 = {};
#if __clang_major__ >= 23
  i32x8 z8 = {};
  __builtin_amdgcn_tensor_load_to_lds(g0.v, g1.v, z4, z4, z8, 0);
#else
  __builtin_amdgcn_tensor_load_to_lds(g0.v, g1.v, z4, z4, 0);
#endif
}
#endif  // USE_TDM

// ---------------------------------------------------------------------------
// f32 -> f16 conversion (grid-stride); plain and transposing variants.
// Weights are stored transposed ([N][K]) so GEMM B-tiles stage with b128 ops.
// ---------------------------------------------------------------------------
__global__ __launch_bounds__(256) void f32_to_f16_kernel(
    const float* __restrict__ in, half_t* __restrict__ out, int n) {
  int i = blockIdx.x * blockDim.x + threadIdx.x;
  int stride = gridDim.x * blockDim.x;
  for (; i < n; i += stride) out[i] = (half_t)in[i];
}

__global__ __launch_bounds__(256) void f32_to_f16_tr_kernel(
    const float* __restrict__ in, half_t* __restrict__ out, int K, int N) {
  // in: [K][N] f32, out: [N][K] f16
  int i = blockIdx.x * blockDim.x + threadIdx.x;
  int stride = gridDim.x * blockDim.x;
  int total = K * N;
  for (; i < total; i += stride) {
    int k = i / N, n = i - k * N;
    out[(size_t)n * K + k] = (half_t)in[i];
  }
}

// ---------------------------------------------------------------------------
// Tiled WMMA GEMM:  C[M,N] = A[M,K] (f16 row-major) x B (f16, given as
// Bt[N][K]).  Block tile 128x128, 8 waves (4x2), wave tile 32x64, K-step 32.
// ---------------------------------------------------------------------------
#define TM 128
#define TN 128
#define TK 32
#define LDA 40   // 32 + 8 halves pad (16B-aligned chunks, bank spread)
#define LDB 40

enum { EPI_QKV = 0, EPI_F32 = 1, EPI_F16 = 2, EPI_GEGLU = 3 };

template <int EPI>
__global__ __launch_bounds__(256) void gemm_kernel(
    const half_t* __restrict__ A, const half_t* __restrict__ Bt,
    const float* __restrict__ bias, void* __restrict__ outp,
    const half_t* __restrict__ aux, int M, int N, int K) {
  __shared__ __align__(16) half_t As[TM * LDA];
  __shared__ __align__(16) half_t Bs[TN * LDB];  // [n][k]

  const int tid  = threadIdx.x;
  const int lane = tid & 31;
  const int wave = tid >> 5;
  const int wm   = wave >> 1;   // 0..3
  const int wn   = wave & 1;    // 0..1
  const int bM   = blockIdx.y * TM;
  const int bN   = blockIdx.x * TN;

  v8f zero = {};
  v8f acc[2][4];
#pragma unroll
  for (int i = 0; i < 2; ++i)
#pragma unroll
    for (int j = 0; j < 4; ++j) acc[i][j] = zero;

  const int sr = tid >> 1;            // 0..127 staging row
  const int sc = (tid & 1) * 16;      // 0 / 16 staging col

  for (int k0 = 0; k0 < K; k0 += TK) {
#ifdef USE_TDM
    if (tid < 32) {  // wave 0 drives the Tensor Data Mover for both tiles
      tdm_load_2d((unsigned)(size_t)(void*)As,
                  (unsigned long long)(size_t)(A + (size_t)bM * K + k0),
                  (unsigned)K, (unsigned)M, (unsigned long long)K, TK, TM);
      tdm_load_2d((unsigned)(size_t)(void*)Bs,
                  (unsigned long long)(size_t)(Bt + (size_t)bN * K + k0),
                  (unsigned)K, (unsigned)N, (unsigned long long)K, TK, TN);
      WAIT_TENSOR();
    }
#else
    {
      const half_t* ga = A + (size_t)(bM + sr) * K + k0 + sc;
      *(v8h*)&As[sr * LDA + sc]     = *(const v8h*)(ga);
      *(v8h*)&As[sr * LDA + sc + 8] = *(const v8h*)(ga + 8);
      const half_t* gb = Bt + (size_t)(bN + sr) * K + k0 + sc;
      *(v8h*)&Bs[sr * LDB + sc]     = *(const v8h*)(gb);
      *(v8h*)&Bs[sr * LDB + sc + 8] = *(const v8h*)(gb + 8);
      if (k0 + TK < K) {  // hint next tiles into cache (global_prefetch_b8)
        __builtin_prefetch(ga + TK, 0, 1);
        __builtin_prefetch(gb + TK, 0, 1);
      }
    }
#endif
    __syncthreads();

    // fragments + WMMA
    const int aK0 = (lane < 16) ? 0 : 8;    // A-layout chunk base
    v16h af[2];
#pragma unroll
    for (int mi = 0; mi < 2; ++mi) {
      const half_t* pa = &As[(wm * 32 + mi * 16 + (lane & 15)) * LDA];
      af[mi] = ld_frag(pa + aK0, pa + aK0 + 16);
    }
    const int bK0 = (lane < 16) ? 0 : 8;    // B-layout chunk base
#pragma unroll
    for (int ni = 0; ni < 4; ++ni) {
      const half_t* pb = &Bs[(wn * 64 + ni * 16 + (lane & 15)) * LDB + bK0];
      v16h bf = ld_frag(pb, pb + 16);
#pragma unroll
      for (int mi = 0; mi < 2; ++mi)
        acc[mi][ni] = wmma_f16(af[mi], bf, acc[mi][ni]);
    }
    __syncthreads();
  }

  // epilogue (C layout: vgpr i -> row i (+8 upper half-wave), col = lane&15)
#pragma unroll
  for (int mi = 0; mi < 2; ++mi)
#pragma unroll
    for (int ni = 0; ni < 4; ++ni)
#pragma unroll
      for (int i = 0; i < 8; ++i) {
        int row = bM + wm * 32 + mi * 16 + i + ((lane >= 16) ? 8 : 0);
        int col = bN + wn * 64 + ni * 16 + (lane & 15);
        float c = acc[mi][ni][i] + bias[col];
        if (EPI == EPI_QKV) {
          int b = row >> 11, s = row & (CS - 1);
          int h = col >> 6, d = col & (CHD - 1);
          ((half_t*)outp)[(((size_t)(b * CH + h)) * CS + s) * CHD + d] =
              (half_t)c;
        } else if (EPI == EPI_F32) {
          ((float*)outp)[(size_t)row * N + col] = c;
        } else if (EPI == EPI_F16) {
          ((half_t*)outp)[(size_t)row * N + col] = (half_t)c;
        } else {  // EPI_GEGLU: out = gelu(aux) * c (exact gelu)
          size_t idx = (size_t)row * N + col;
          float g = (float)aux[idx];
          float gl = 0.5f * g * (1.0f + erff(g * 0.70710678118654752f));
          ((half_t*)outp)[idx] = (half_t)(gl * c);
        }
      }
}

// ---------------------------------------------------------------------------
// Flash attention: block = one (b, h, 64-query block); 4 waves, each owns
// 16 query rows.  Streams K/V in 64-key tiles, online softmax in LDS.
// q/k/v layout: [B, H, S, HD] f16; Q pre-scaled by 1/sqrt(HD) at load.
// ---------------------------------------------------------------------------
__global__ __launch_bounds__(128) void flash_attn_kernel(
    const half_t* __restrict__ Q, const half_t* __restrict__ Km,
    const half_t* __restrict__ V, half_t* __restrict__ ctx) {
  const int qb = blockIdx.x;  // 0..31
  const int h  = blockIdx.y;  // 0..7
  const int b  = blockIdx.z;  // 0..3
  const int tid  = threadIdx.x;
  const int lane = tid & 31;
  const int wave = tid >> 5;  // 0..3

  __shared__ __align__(16) half_t Qs[64 * 72];
  __shared__ __align__(16) half_t Ks[64 * 72];   // [key][d]
  __shared__ __align__(16) half_t Vt[64 * 72];   // [d][key]
  __shared__ __align__(16) half_t Pb[64 * 72];   // probs [q][key]
  __shared__ float Sb[64 * 64];
  __shared__ float mrow[64], lrow[64], frow[64];

  const size_t bh = ((size_t)(b * CH + h)) * CS * CHD;
  {  // stage Q (scaled by 1/sqrt(64) = 0.125)
    const int r = tid >> 1, c = (tid & 1) * 32;
    const half_t* gq = Q + bh + (size_t)(qb * 64 + r) * CHD + c;
#pragma unroll
    for (int j = 0; j < 32; ++j)
      Qs[r * 72 + c + j] = (half_t)((float)gq[j] * 0.125f);
  }
  if (tid < 64) { mrow[tid] = -1e30f; lrow[tid] = 0.0f; }

  v8f zero = {};
  v8f oacc[4];
#pragma unroll
  for (int i = 0; i < 4; ++i) oacc[i] = zero;

  const int qr = wave * 16;

  for (int kb = 0; kb < CS / 64; ++kb) {
    __syncthreads();
    {  // stage K [key][d], V transposed [d][key]
      const int r = tid >> 1, c = (tid & 1) * 32;
      const half_t* gk = Km + bh + (size_t)(kb * 64 + r) * CHD + c;
#pragma unroll
      for (int j = 0; j < 32; j += 8)
        *(v8h*)&Ks[r * 72 + c + j] = *(const v8h*)(gk + j);
      const half_t* gv = V + bh + (size_t)(kb * 64 + r) * CHD + c;
#pragma unroll
      for (int j = 0; j < 32; ++j) Vt[(c + j) * 72 + r] = gv[j];
    }
    __syncthreads();

    // S = Qs x Ks^T : 16 rows x 64 keys per wave
    v8f sacc[4];
#pragma unroll
    for (int i = 0; i < 4; ++i) sacc[i] = zero;
#pragma unroll
    for (int ks = 0; ks < CHD; ks += 32) {
      const int aK0 = ks + ((lane < 16) ? 0 : 8);
      const half_t* pa = &Qs[(qr + (lane & 15)) * 72];
      v16h af = ld_frag(pa + aK0, pa + aK0 + 16);
      const int bK0 = ks + ((lane < 16) ? 0 : 8);
#pragma unroll
      for (int ni = 0; ni < 4; ++ni) {
        const half_t* pb = &Ks[(ni * 16 + (lane & 15)) * 72 + bK0];
        v16h bf = ld_frag(pb, pb + 16);
        sacc[ni] = wmma_f16(af, bf, sacc[ni]);
      }
    }
#pragma unroll
    for (int ni = 0; ni < 4; ++ni)
#pragma unroll
      for (int i = 0; i < 8; ++i) {
        int row = qr + i + ((lane >= 16) ? 8 : 0);
        int col = ni * 16 + (lane & 15);
        Sb[row * 64 + col] = sacc[ni][i];
      }
    __syncthreads();

    // online softmax: one thread per query row
    if (tid < 64) {
      float mold = mrow[tid];
      float mx = mold;
#pragma unroll 8
      for (int j = 0; j < 64; ++j) mx = fmaxf(mx, Sb[tid * 64 + j]);
      float fac = __expf(mold - mx);
      float sum = 0.0f;
#pragma unroll 8
      for (int j = 0; j < 64; ++j) {
        float p = __expf(Sb[tid * 64 + j] - mx);
        Pb[tid * 72 + j] = (half_t)p;
        sum += p;
      }
      mrow[tid] = mx;
      lrow[tid] = lrow[tid] * fac + sum;
      frow[tid] = fac;
    }
    __syncthreads();

    // rescale O, then O += P x V
#pragma unroll
    for (int ni = 0; ni < 4; ++ni)
#pragma unroll
      for (int i = 0; i < 8; ++i) {
        int row = qr + i + ((lane >= 16) ? 8 : 0);
        oacc[ni][i] *= frow[row];
      }
#pragma unroll
    for (int ks = 0; ks < 64; ks += 32) {
      const int aK0 = ks + ((lane < 16) ? 0 : 8);
      const half_t* pa = &Pb[(qr + (lane & 15)) * 72];
      v16h af = ld_frag(pa + aK0, pa + aK0 + 16);
      const int bK0 = ks + ((lane < 16) ? 0 : 8);
#pragma unroll
      for (int ni = 0; ni < 4; ++ni) {
        const half_t* pb = &Vt[(ni * 16 + (lane & 15)) * 72 + bK0];
        v16h bf = ld_frag(pb, pb + 16);
        oacc[ni] = wmma_f16(af, bf, oacc[ni]);
      }
    }
  }
  __syncthreads();

  // write ctx in [B*S, D] row-major f16 (ready for Wo GEMM)
#pragma unroll
  for (int ni = 0; ni < 4; ++ni)
#pragma unroll
    for (int i = 0; i < 8; ++i) {
      int row = qr + i + ((lane >= 16) ? 8 : 0);
      int col = ni * 16 + (lane & 15);
      float o = oacc[ni][i] / lrow[row];
      int s = qb * 64 + row;
      ctx[((size_t)b * CS + s) * CD + h * CHD + col] = (half_t)o;
    }
}

// ---------------------------------------------------------------------------
// Fused residual + LayerNorm over D=512; optional f16 copy for next GEMM.
// ---------------------------------------------------------------------------
template <bool WRITE_H>
__global__ __launch_bounds__(128) void resln_kernel(
    const float* __restrict__ X, const float* __restrict__ Y,
    const float* __restrict__ g, const float* __restrict__ be,
    float* __restrict__ outf, half_t* __restrict__ outh) {
  const int row = blockIdx.x;
  const int tid = threadIdx.x;
  __shared__ float red[128];

  const float* xr = X + (size_t)row * CD;
  const float* yr = Y + (size_t)row * CD;
  float v[4];
  float s = 0.0f;
#pragma unroll
  for (int i = 0; i < 4; ++i) {
    v[i] = xr[tid + i * 128] + yr[tid + i * 128];
    s += v[i];
  }
  red[tid] = s;
  __syncthreads();
  for (int off = 64; off > 0; off >>= 1) {
    if (tid < off) red[tid] += red[tid + off];
    __syncthreads();
  }
  float mu = red[0] * (1.0f / CD);
  __syncthreads();

  float vs = 0.0f;
#pragma unroll
  for (int i = 0; i < 4; ++i) {
    float d = v[i] - mu;
    vs += d * d;
  }
  red[tid] = vs;
  __syncthreads();
  for (int off = 64; off > 0; off >>= 1) {
    if (tid < off) red[tid] += red[tid + off];
    __syncthreads();
  }
  float rstd = rsqrtf(red[0] * (1.0f / CD) + 1e-5f);

#pragma unroll
  for (int i = 0; i < 4; ++i) {
    int c = tid + i * 128;
    float o = (v[i] - mu) * rstd * g[c] + be[c];
    outf[(size_t)row * CD + c] = o;
    if (WRITE_H) outh[(size_t)row * CD + c] = (half_t)o;
  }
}

// ---------------------------------------------------------------------------
// Host orchestration
// ---------------------------------------------------------------------------
static inline void conv16(const float* in, half_t* out, int n, hipStream_t s) {
  int blocks = (n + 255) / 256;
  if (blocks > 8192) blocks = 8192;
  f32_to_f16_kernel<<<blocks, 256, 0, s>>>(in, out, n);
}

static inline void conv16t(const float* in, half_t* out, int K, int N,
                           hipStream_t s) {
  int n = K * N;
  int blocks = (n + 255) / 256;
  if (blocks > 8192) blocks = 8192;
  f32_to_f16_tr_kernel<<<blocks, 256, 0, s>>>(in, out, K, N);
}

extern "C" void kernel_launch(void* const* d_in, const int* in_sizes, int n_in,
                              void* d_out, int out_size, void* d_ws,
                              size_t ws_size, hipStream_t stream) {
  (void)in_sizes; (void)n_in; (void)out_size; (void)ws_size;

  const float* x    = (const float*)d_in[0];
  const float* Wq   = (const float*)d_in[1];
  const float* bq   = (const float*)d_in[2];
  const float* Wk   = (const float*)d_in[3];
  const float* bk   = (const float*)d_in[4];
  const float* Wv   = (const float*)d_in[5];
  const float* bv   = (const float*)d_in[6];
  const float* Wo   = (const float*)d_in[7];
  const float* bo   = (const float*)d_in[8];
  const float* Wff  = (const float*)d_in[9];
  const float* bff  = (const float*)d_in[10];
  const float* Vff  = (const float*)d_in[11];
  const float* bvff = (const float*)d_in[12];
  const float* W2ff = (const float*)d_in[13];
  const float* b2ff = (const float*)d_in[14];
  const float* g1   = (const float*)d_in[15];
  const float* be1  = (const float*)d_in[16];
  const float* g2   = (const float*)d_in[17];
  const float* be2  = (const float*)d_in[18];
  float* out = (float*)d_out;

  char* ws = (char*)d_ws;
  size_t off = 0;
  auto alloc = [&](size_t bytes) -> void* {
    void* p = ws + off;
    off += (bytes + 255) & ~(size_t)255;
    return p;
  };

  half_t* xh   = (half_t*)alloc((size_t)CM * CD * 2);
  half_t* WqT  = (half_t*)alloc((size_t)CD * CD * 2);
  half_t* WkT  = (half_t*)alloc((size_t)CD * CD * 2);
  half_t* WvT  = (half_t*)alloc((size_t)CD * CD * 2);
  half_t* WoT  = (half_t*)alloc((size_t)CD * CD * 2);
  half_t* WffT = (half_t*)alloc((size_t)CD * CF * 2);
  half_t* VffT = (half_t*)alloc((size_t)CD * CF * 2);
  half_t* W2T  = (half_t*)alloc((size_t)CF * CD * 2);
  half_t* qh   = (half_t*)alloc((size_t)CM * CD * 2);
  half_t* kh   = (half_t*)alloc((size_t)CM * CD * 2);
  half_t* vh   = (half_t*)alloc((size_t)CM * CD * 2);
  half_t* ctxh = (half_t*)alloc((size_t)CM * CD * 2);
  half_t* x1h  = (half_t*)alloc((size_t)CM * CD * 2);
  half_t* gbuf = (half_t*)alloc((size_t)CM * CF * 2);
  half_t* hbuf = (half_t*)alloc((size_t)CM * CF * 2);
  float*  aout = (float*)alloc((size_t)CM * CD * 4);
  float*  x1f  = (float*)alloc((size_t)CM * CD * 4);
  float*  fout = (float*)alloc((size_t)CM * CD * 4);

  // 0) fp32 -> fp16 (weights transposed to [N][K] for the GEMM B path)
  conv16(x, xh, CM * CD, stream);
  conv16t(Wq, WqT, CD, CD, stream);
  conv16t(Wk, WkT, CD, CD, stream);
  conv16t(Wv, WvT, CD, CD, stream);
  conv16t(Wo, WoT, CD, CD, stream);
  conv16t(Wff, WffT, CD, CF, stream);
  conv16t(Vff, VffT, CD, CF, stream);
  conv16t(W2ff, W2T, CF, CD, stream);

  dim3 blk(256);
  // 1) QKV projections -> [B,H,S,HD] f16
  {
    dim3 grd(CD / TN, CM / TM);
    gemm_kernel<EPI_QKV><<<grd, blk, 0, stream>>>(xh, WqT, bq, qh, nullptr,
                                                  CM, CD, CD);
    gemm_kernel<EPI_QKV><<<grd, blk, 0, stream>>>(xh, WkT, bk, kh, nullptr,
                                                  CM, CD, CD);
    gemm_kernel<EPI_QKV><<<grd, blk, 0, stream>>>(xh, WvT, bv, vh, nullptr,
                                                  CM, CD, CD);
  }
  // 2) flash attention -> ctx f16 [B*S, D]
  flash_attn_kernel<<<dim3(CS / 64, CH, CB), dim3(128), 0, stream>>>(qh, kh, vh,
                                                                     ctxh);
  // 3) output projection -> attn_out f32
  gemm_kernel<EPI_F32><<<dim3(CD / TN, CM / TM), blk, 0, stream>>>(
      ctxh, WoT, bo, aout, nullptr, CM, CD, CD);
  // 4) x1 = LN(x + attn_out)
  resln_kernel<true><<<CM, 128, 0, stream>>>(x, aout, g1, be1, x1f, x1h);
  // 5) GEGLU FFN
  gemm_kernel<EPI_F16><<<dim3(CF / TN, CM / TM), blk, 0, stream>>>(
      x1h, WffT, bff, gbuf, nullptr, CM, CF, CD);
  gemm_kernel<EPI_GEGLU><<<dim3(CF / TN, CM / TM), blk, 0, stream>>>(
      x1h, VffT, bvff, hbuf, gbuf, CM, CF, CD);
  gemm_kernel<EPI_F32><<<dim3(CD / TN, CM / TM), blk, 0, stream>>>(
      hbuf, W2T, b2ff, fout, nullptr, CM, CD, CF);
  // 6) out = LN(x1 + ff_out)
  resln_kernel<false><<<CM, 128, 0, stream>>>(x1f, fout, g2, be2, out, nullptr);
}
